// MultiheadAttention_36034775613914
// MI455X (gfx1250) — matrix operand order repaired and checked by reading, not compiled, not run on hardware
//
#include <hip/hip_runtime.h>

#define EMB   1024
#define NH    16
#define HD    64
#define SEQ   2048
#define BATCH 4
#define MTOT  (BATCH*SEQ)   // 8192 rows

typedef __attribute__((ext_vector_type(16))) __bf16          v16bf;
typedef __attribute__((ext_vector_type(8)))  float           v8f;
typedef __attribute__((ext_vector_type(16))) unsigned short  v16us;
typedef __attribute__((ext_vector_type(8)))  unsigned short  v8us;

// ---------- helpers ----------
__device__ __forceinline__ unsigned short f2bf(float f) {
  unsigned u = __builtin_bit_cast(unsigned, f);
  u += 0x7FFFu + ((u >> 16) & 1u);          // round-to-nearest-even
  return (unsigned short)(u >> 16);
}

__device__ __forceinline__ v8f wmma_bf16(v16bf a, v16bf b, v8f c) {
  // v_wmma_f32_16x16x32_bf16: D = A(16x32) * B(32x16) + C
  return __builtin_amdgcn_wmma_f32_16x16x32_bf16(
      false, a, false, b, (short)0, c, false, false);
}

// A fragment (16x32) from fp32 row-major source (fallback path)
// ISA layout: lane<16 -> M=lane, K{0..7,16..23}; lane>=16 -> M=lane-16, K{8..15,24..31}
__device__ __forceinline__ v16bf load_a_f32(const float* base, int ld, int k0) {
  int lane = threadIdx.x & 31;
  int r = lane & 15, h = lane >> 4;
  const float* p = base + (size_t)r * ld + k0 + h * 8;
  v16us t;
#pragma unroll
  for (int i = 0; i < 8; i++) { t[i] = f2bf(p[i]); t[i + 8] = f2bf(p[16 + i]); }
  return __builtin_bit_cast(v16bf, t);
}

// A fragment from bf16 row-major source (2 x 16B loads, no conversion)
__device__ __forceinline__ v16bf load_a_bf(const unsigned short* base, int ld, int k0) {
  int lane = threadIdx.x & 31;
  int r = lane & 15, h = lane >> 4;
  const unsigned short* p = base + (size_t)r * ld + k0 + h * 8;
  v8us c0 = *(const v8us*)p;
  v8us c1 = *(const v8us*)(p + 16);
  v16us t = __builtin_shufflevector(c0, c1, 0,1,2,3,4,5,6,7,8,9,10,11,12,13,14,15);
  return __builtin_bit_cast(v16bf, t);
}

// B fragment (32x16): row r = column n; two 16B loads per lane
// ISA layout: lane<16 -> N=lane, K{0..7,16..23}; lane>=16 -> N=lane-16, K{8..15,24..31}
__device__ __forceinline__ v16bf load_b_bf(const unsigned short* base, int ld, int k0) {
  int lane = threadIdx.x & 31;
  int r = lane & 15, h = lane >> 4;
  const unsigned short* p = base + (size_t)r * ld + k0 + h * 8;
  v8us c0 = *(const v8us*)p;
  v8us c1 = *(const v8us*)(p + 16);
  v16us t = __builtin_shufflevector(c0, c1, 0,1,2,3,4,5,6,7,8,9,10,11,12,13,14,15);
  return __builtin_bit_cast(v16bf, t);
}

// reductions across 16 lanes (xor masks < 16 stay within each half-wave,
// matching C-fragment row ownership: lanes0-15 <-> rows v, lanes16-31 <-> rows v+8)
__device__ __forceinline__ float hmax16(float v) {
#pragma unroll
  for (int m = 1; m < 16; m <<= 1) v = fmaxf(v, __shfl_xor(v, m, 32));
  return v;
}
__device__ __forceinline__ float hsum16(float v) {
#pragma unroll
  for (int m = 1; m < 16; m <<= 1) v += __shfl_xor(v, m, 32);
  return v;
}

// ---------- kernel 0: one-shot fp32 -> bf16 conversion (bandwidth-trivial) ----------
__global__ void __launch_bounds__(256)
cvt_kernel(const float* __restrict__ src, unsigned short* __restrict__ dst, int n) {
  int i = (blockIdx.x * 256 + threadIdx.x) * 8;
  if (i >= n) return;
  float4 f0 = *(const float4*)(src + i);
  float4 f1 = *(const float4*)(src + i + 4);
  v8us o;
  o[0] = f2bf(f0.x); o[1] = f2bf(f0.y); o[2] = f2bf(f0.z); o[3] = f2bf(f0.w);
  o[4] = f2bf(f1.x); o[5] = f2bf(f1.y); o[6] = f2bf(f1.z); o[7] = f2bf(f1.w);
  *(v8us*)(dst + i) = o;
}

// ---------- kernel 1: QKV projection (32x64 tile per wave; 8 WMMA / 6 loads) ----------
template <bool XBF>
__global__ void __launch_bounds__(256)
qkv_kernel(const float* __restrict__ x, const unsigned short* __restrict__ xbf,
           const unsigned short* __restrict__ Wq, const float* __restrict__ bqkv,
           unsigned short* Qb, unsigned short* Kb, unsigned short* Vt) {
  int wv = threadIdx.x >> 5, lane = threadIdx.x & 31;
  int waveId = blockIdx.x * 8 + wv;
  int mt = waveId / 48, nt = waveId % 48;            // 256 m32-tiles x 48 n64-tiles
  int m0 = mt * 32, n0 = nt * 64;

  v8f acc[2][4] = {};
  for (int k0 = 0; k0 < EMB; k0 += 32) {
    v16bf a0, a1;
    if (XBF) {
      a0 = load_a_bf(xbf + (size_t)m0 * EMB, EMB, k0);
      a1 = load_a_bf(xbf + (size_t)(m0 + 16) * EMB, EMB, k0);
    } else {
      a0 = load_a_f32(x + (size_t)m0 * EMB, EMB, k0);
      a1 = load_a_f32(x + (size_t)(m0 + 16) * EMB, EMB, k0);
    }
#pragma unroll
    for (int j = 0; j < 4; j++) {
      v16bf b = load_b_bf(Wq + (size_t)(n0 + j * 16) * EMB, EMB, k0);
      acc[0][j] = wmma_bf16(a0, b, acc[0][j]);
      acc[1][j] = wmma_bf16(a1, b, acc[1][j]);
    }
  }

  int half = lane >> 4, ln = lane & 15;
  int which = n0 >> 10;                 // 0=Q 1=K 2=V (wave-uniform)
  int h = (n0 & 1023) >> 6;             // head
  int b_ = m0 >> 11;                    // batch (32 | 2048, tiles never straddle)
  int s0 = m0 & 2047;
  size_t bh = (size_t)b_ * NH + h;
#pragma unroll
  for (int u = 0; u < 2; u++) {
#pragma unroll
    for (int j = 0; j < 4; j++) {
      int n = n0 + j * 16 + ln;
      int d = j * 16 + ln;
      float bias = bqkv[n];
#pragma unroll
      for (int v = 0; v < 8; v++) {
        int s = s0 + u * 16 + v + half * 8;
        unsigned short bf = f2bf(acc[u][j][v] + bias);
        if (which == 0)      Qb[(bh * SEQ + s) * HD + d] = bf;   // [B,H,S,D]
        else if (which == 1) Kb[(bh * SEQ + s) * HD + d] = bf;   // [B,H,S,D]
        else                 Vt[(bh * HD + d) * SEQ + s] = bf;   // [B,H,D,S]
      }
    }
  }
}

// ---------- kernel 2: causal flash attention (32 queries per wave) ----------
__global__ void __launch_bounds__(256)
attn_kernel(unsigned short* QO, const unsigned short* __restrict__ Kb,
            const unsigned short* __restrict__ Vt) {
  __shared__ __align__(16) unsigned short Pst[8][1024];   // 2KB per wave (2 P tiles)
  int wv = threadIdx.x >> 5, lane = threadIdx.x & 31;
  int waveId = blockIdx.x * 8 + wv;
  int qt = waveId & 63;                  // 64 q32-tiles per head
  int bh = waveId >> 6;                  // 0..63 = b*16+h
  int q0 = qt * 32;

  unsigned short* Qbase = QO + ((size_t)bh * SEQ + q0) * HD;
  const unsigned short* Kbase = Kb + (size_t)bh * SEQ * HD;
  const unsigned short* Vbase = Vt + (size_t)bh * HD * SEQ;

  // Q tile (2 row sub-tiles x 2 d-slices) into registers once
  v16bf qa0[2], qa1[2];
#pragma unroll
  for (int u = 0; u < 2; u++) {
    qa0[u] = load_a_bf(Qbase + (size_t)u * 16 * HD, HD, 0);
    qa1[u] = load_a_bf(Qbase + (size_t)u * 16 * HD, HD, 32);
  }

  v8f o0[2] = {}, o1[2] = {}, o2[2] = {}, o3[2] = {};
  float mrow[2][8], lrow[2][8];
#pragma unroll
  for (int u = 0; u < 2; u++)
#pragma unroll
    for (int v = 0; v < 8; v++) { mrow[u][v] = -1e30f; lrow[u][v] = 0.f; }

  int half = lane >> 4, ln = lane & 15;
  unsigned short* myP = &Pst[wv][0];
  const float LOG2E = 1.4426950408889634f;

  for (int t0 = 0; t0 <= q0 + 31; t0 += 32) {
    // K fragments shared by both query sub-tiles
    v16bf kb00 = load_b_bf(Kbase + (size_t)t0 * HD, HD, 0);
    v16bf kb01 = load_b_bf(Kbase + (size_t)t0 * HD, HD, 32);
    v16bf kb10 = load_b_bf(Kbase + (size_t)(t0 + 16) * HD, HD, 0);
    v16bf kb11 = load_b_bf(Kbase + (size_t)(t0 + 16) * HD, HD, 32);

    v16bf pf[2];
    int tc0 = t0 + ln, tc1 = t0 + 16 + ln;
#pragma unroll
    for (int u = 0; u < 2; u++) {
      v8f s0 = {}, s1 = {};
      s0 = wmma_bf16(qa0[u], kb00, s0);
      s0 = wmma_bf16(qa1[u], kb01, s0);
      s1 = wmma_bf16(qa0[u], kb10, s1);
      s1 = wmma_bf16(qa1[u], kb11, s1);

      // online softmax, branch-free causal mask
      float p0[8], p1[8], alpha[8];
#pragma unroll
      for (int v = 0; v < 8; v++) {
        int qrow = q0 + u * 16 + v + half * 8;
        float a = (tc0 <= qrow) ? s0[v] * 0.125f : -1e30f;   // 1/sqrt(64)
        float b = (tc1 <= qrow) ? s1[v] * 0.125f : -1e30f;
        float mx = hmax16(fmaxf(a, b));
        float mn = fmaxf(mrow[u][v], mx);
        float e0 = exp2f((a - mn) * LOG2E);
        float e1 = exp2f((b - mn) * LOG2E);
        float al = exp2f((mrow[u][v] - mn) * LOG2E);
        lrow[u][v] = lrow[u][v] * al + hsum16(e0 + e1);
        mrow[u][v] = mn;
        alpha[v] = al;
        p0[v] = e0; p1[v] = e1;
      }
#pragma unroll
      for (int v = 0; v < 8; v++) {
        o0[u][v] = o0[u][v] * alpha[v]; o1[u][v] = o1[u][v] * alpha[v];
        o2[u][v] = o2[u][v] * alpha[v]; o3[u][v] = o3[u][v] * alpha[v];
      }

      // C-layout -> A-layout for P via per-wave LDS bounce
      unsigned short* tP = myP + u * 512;
#pragma unroll
      for (int v = 0; v < 8; v++) {
        int row = v + half * 8;
        tP[row * 32 + ln]      = f2bf(p0[v]);
        tP[row * 32 + 16 + ln] = f2bf(p1[v]);
      }
      asm volatile("s_wait_dscnt 0" ::: "memory");
      const unsigned short* pp = tP + (size_t)ln * 32 + half * 8;
      v8us c0 = *(const v8us*)pp;
      v8us c1 = *(const v8us*)(pp + 16);
      pf[u] = __builtin_bit_cast(
          v16bf, __builtin_shufflevector(c0, c1, 0,1,2,3,4,5,6,7,8,9,10,11,12,13,14,15));
    }

    // V fragments shared by both query sub-tiles: O += P @ V
    v16bf vb0 = load_b_bf(Vbase,            SEQ, t0);
    v16bf vb1 = load_b_bf(Vbase + 16 * SEQ, SEQ, t0);
    v16bf vb2 = load_b_bf(Vbase + 32 * SEQ, SEQ, t0);
    v16bf vb3 = load_b_bf(Vbase + 48 * SEQ, SEQ, t0);
#pragma unroll
    for (int u = 0; u < 2; u++) {
      o0[u] = wmma_bf16(pf[u], vb0, o0[u]);
      o1[u] = wmma_bf16(pf[u], vb1, o1[u]);
      o2[u] = wmma_bf16(pf[u], vb2, o2[u]);
      o3[u] = wmma_bf16(pf[u], vb3, o3[u]);
    }
  }

  // normalize and store O in-place over the (now dead) Q tile, bf16 [B,H,S,D]
#pragma unroll
  for (int u = 0; u < 2; u++) {
#pragma unroll
    for (int v = 0; v < 8; v++) {
      float inv = 1.0f / lrow[u][v];
      unsigned short* orow = Qbase + (size_t)(u * 16 + v + half * 8) * HD;
      orow[ln]      = f2bf(o0[u][v] * inv);
      orow[16 + ln] = f2bf(o1[u][v] * inv);
      orow[32 + ln] = f2bf(o2[u][v] * inv);
      orow[48 + ln] = f2bf(o3[u][v] * inv);
    }
  }
}

// ---------- kernel 3: output projection (32x64 tile per wave) ----------
__global__ void __launch_bounds__(256)
proj_kernel(const unsigned short* __restrict__ O, const unsigned short* __restrict__ Wp,
            const float* __restrict__ bp, float* __restrict__ out) {
  int wv = threadIdx.x >> 5, lane = threadIdx.x & 31;
  int waveId = blockIdx.x * 8 + wv;
  int mt = waveId >> 4, nt = waveId & 15;            // 256 m32-tiles x 16 n64-tiles
  int m0 = mt * 32, n0 = nt * 64;
  int half = lane >> 4, ln = lane & 15;
  int b_ = m0 >> 11, s0 = m0 & 2047;

  v8f acc[2][4] = {};
  for (int k0 = 0; k0 < EMB; k0 += 32) {
    int h = k0 >> 6, d0 = k0 & 63;                   // k-chunks never cross a head
    const unsigned short* Ab = O + (((size_t)b_ * NH + h) * SEQ + s0) * HD + d0;
    v16bf a0 = load_a_bf(Ab, HD, 0);
    v16bf a1 = load_a_bf(Ab + (size_t)16 * HD, HD, 0);
#pragma unroll
    for (int j = 0; j < 4; j++) {
      v16bf b = load_b_bf(Wp + (size_t)(n0 + j * 16) * EMB, EMB, k0);
      acc[0][j] = wmma_bf16(a0, b, acc[0][j]);
      acc[1][j] = wmma_bf16(a1, b, acc[1][j]);
    }
  }
#pragma unroll
  for (int u = 0; u < 2; u++) {
#pragma unroll
    for (int j = 0; j < 4; j++) {
      int n = n0 + j * 16 + ln;
      float bias = bp[n];
#pragma unroll
      for (int v = 0; v < 8; v++) {
        int m = m0 + u * 16 + v + half * 8;
        out[(size_t)m * EMB + n] = acc[u][j][v] + bias;
      }
    }
  }
}

// ---------- launch ----------
extern "C" void kernel_launch(void* const* d_in, const int* in_sizes, int n_in,
                              void* d_out, int out_size, void* d_ws, size_t ws_size,
                              hipStream_t stream) {
  const float* x    = (const float*)d_in[0];
  const float* Wqkv = (const float*)d_in[1];
  const float* bqkv = (const float*)d_in[2];
  const float* Wp   = (const float*)d_in[3];
  const float* bp   = (const float*)d_in[4];
  float* out = (float*)d_out;

  // workspace layout (bf16 element offsets)
  unsigned short* ws = (unsigned short*)d_ws;
  const size_t QO_OFF  = 0;                         // 8M elems (Q, then O in-place)
  const size_t K_OFF   = (size_t)MTOT * EMB;        // 8M elems
  const size_t V_OFF   = (size_t)2 * MTOT * EMB;    // 8M elems (V transposed)
  const size_t WQ_OFF  = (size_t)3 * MTOT * EMB;    // 3M elems (W_qkv bf16)
  const size_t WP_OFF  = WQ_OFF + (size_t)3 * EMB * EMB;   // 1M elems (W_proj bf16)
  const size_t XBF_OFF = WP_OFF + (size_t)EMB * EMB;       // 8M elems (x bf16, optional)
  const size_t NEED_FULL = (XBF_OFF + (size_t)MTOT * EMB) * 2;   // 72 MiB

  unsigned short* Qb  = ws + QO_OFF;
  unsigned short* Kb  = ws + K_OFF;
  unsigned short* Vt  = ws + V_OFF;
  unsigned short* Wqb = ws + WQ_OFF;
  unsigned short* Wpb = ws + WP_OFF;
  unsigned short* xbf = ws + XBF_OFF;

  const bool use_xbf = (ws_size >= NEED_FULL);      // constant across replays

  // one-shot fp32->bf16 conversions (hoists conversion VALU out of WMMA loops)
  cvt_kernel<<<(3 * EMB * EMB) / 2048, 256, 0, stream>>>(Wqkv, Wqb, 3 * EMB * EMB);
  cvt_kernel<<<(EMB * EMB) / 2048, 256, 0, stream>>>(Wp, Wpb, EMB * EMB);
  if (use_xbf)
    cvt_kernel<<<(MTOT * EMB) / 2048, 256, 0, stream>>>(x, xbf, MTOT * EMB);

  if (use_xbf)
    qkv_kernel<true><<<1536, 256, 0, stream>>>(x, xbf, Wqb, bqkv, Qb, Kb, Vt);
  else
    qkv_kernel<false><<<1536, 256, 0, stream>>>(x, xbf, Wqb, bqkv, Qb, Kb, Vt);

  attn_kernel<<<512, 256, 0, stream>>>(Qb, Kb, Vt);        // 4096 waves
  proj_kernel<<<512, 256, 0, stream>>>(Qb, Wpb, bp, out);  // 4096 waves
}
